// AutoEncoder_37374805410270
// MI455X (gfx1250) — matrix operand order repaired and checked by reading, not compiled, not run on hardware
//
#include <hip/hip_runtime.h>

// ---------------- problem constants ----------------
constexpr int Bn   = 4096;      // batch rows
constexpr int En   = 2048;      // embed dim (K of encoder GEMM)
constexpr int Fn   = 65536;     // features (N of encoder GEMM)
constexpr int TOPK = 64;

// ---------------- GEMM / top-k tiling ----------------
constexpr int MB     = 256;     // rows per block
constexpr int FT     = 2048;    // feature columns per block (top-k chunk)
constexpr int NSUB   = 256;     // columns per sub-tile
constexpr int KT     = 32;      // K per WMMA panel step
constexpr int XSTR   = 40;      // padded LDS row stride (bf16 elems, 16B aligned)
constexpr int NCHUNK = Fn / FT; // 32 partial chunks per row
constexpr int CAND   = NCHUNK * TOPK; // 2048 merge candidates per row

#define NEG_INF (-3.402823466e+38f)

typedef __bf16 v8bf  __attribute__((ext_vector_type(8)));
typedef __bf16 v16bf __attribute__((ext_vector_type(16)));
typedef float  v8f   __attribute__((ext_vector_type(8)));

// LDS layout (bytes) for the GEMM kernel
constexpr int XS_ELEMS   = MB * XSTR;             // one X panel buffer (bf16)
constexpr int WS_ELEMS   = NSUB * XSTR;           // one W panel buffer (bf16)
constexpr int OFF_XS     = 0;
constexpr int OFF_WS     = OFF_XS + 2 * XS_ELEMS * 2;
constexpr int OFF_STAGE  = OFF_WS + 2 * WS_ELEMS * 2;
constexpr int OFF_LV     = OFF_STAGE + MB * 64 * 4;
constexpr int OFF_LI     = OFF_LV + MB * TOPK * 4;
constexpr int OFF_LMIN   = OFF_LI + MB * TOPK * 4;
constexpr int OFF_LPOS   = OFF_LMIN + MB * 4;
constexpr int SMEM_GEMM  = OFF_LPOS + MB * 4;     // ~274 KB (<= 320 KB WGP LDS)
constexpr int SMEM_MERGE = 8 * CAND * 8;          // 128 KB

// ======================================================================
// Kernel 0: x_bf16[b][e] = bf16(embed[b][e] - bias[e])
// ======================================================================
__global__ __launch_bounds__(256) void prep_x_kernel(const float* __restrict__ embed,
                                                     const float* __restrict__ bias,
                                                     __bf16* __restrict__ xbf) {
    int i8 = (blockIdx.x * 256 + threadIdx.x) * 8;
    int e  = i8 & (En - 1);
    const float4* x4 = (const float4*)(embed + i8);
    const float4* b4 = (const float4*)(bias + e);
    float4 x0 = x4[0], x1 = x4[1];
    float4 b0 = b4[0], b1 = b4[1];
    union { __bf16 h[8]; uint4 u; } pk;
    pk.h[0] = (__bf16)(x0.x - b0.x); pk.h[1] = (__bf16)(x0.y - b0.y);
    pk.h[2] = (__bf16)(x0.z - b0.z); pk.h[3] = (__bf16)(x0.w - b0.w);
    pk.h[4] = (__bf16)(x1.x - b1.x); pk.h[5] = (__bf16)(x1.y - b1.y);
    pk.h[6] = (__bf16)(x1.z - b1.z); pk.h[7] = (__bf16)(x1.w - b1.w);
    *(uint4*)(xbf + i8) = pk.u;
}

// ======================================================================
// panel loaders (cooperative, 1024 threads)
// ======================================================================
__device__ __forceinline__ void load_xpanel(const __bf16* __restrict__ xbf,
                                            __bf16* __restrict__ xs,
                                            int m0, int k0, int t) {
    int row = t >> 2, part = t & 3;
    const uint4* src = (const uint4*)(xbf + (size_t)(m0 + row) * En + k0 + part * 8);
    *(uint4*)(xs + row * XSTR + part * 8) = *src;
}

__device__ __forceinline__ void load_wpanel(const float* __restrict__ encw,
                                            __bf16* __restrict__ wsp,
                                            int f0s, int k0, int t) {
    int row = t >> 2, part = t & 3;
    const float4* s = (const float4*)(encw + (size_t)(f0s + row) * En + k0 + part * 8);
    float4 a = s[0], b = s[1];
    union { __bf16 h[8]; uint4 u; } pk;
    pk.h[0] = (__bf16)a.x; pk.h[1] = (__bf16)a.y; pk.h[2] = (__bf16)a.z; pk.h[3] = (__bf16)a.w;
    pk.h[4] = (__bf16)b.x; pk.h[5] = (__bf16)b.y; pk.h[6] = (__bf16)b.z; pk.h[7] = (__bf16)b.w;
    *(uint4*)(wsp + row * XSTR + part * 8) = pk.u;
}

// ======================================================================
// Kernel 1: fused bf16-WMMA GEMM + streaming per-row top-64 (per F-chunk)
//   grid = (Bn/MB = 16, NCHUNK = 32), block = 1024 (32 wave32 waves)
// ======================================================================
__global__ __launch_bounds__(1024, 1) void gemm_topk_kernel(const __bf16* __restrict__ xbf,
                                                            const float* __restrict__ encw,
                                                            float* __restrict__ pv,
                                                            int* __restrict__ pi) {
    extern __shared__ char smem[];
    __bf16* xs    = (__bf16*)(smem + OFF_XS);
    __bf16* wsp   = (__bf16*)(smem + OFF_WS);
    float*  stage = (float*)(smem + OFF_STAGE);
    float*  lv    = (float*)(smem + OFF_LV);
    int*    li    = (int*)(smem + OFF_LI);
    float*  lmin  = (float*)(smem + OFF_LMIN);
    int*    lpos  = (int*)(smem + OFF_LPOS);

    const int t    = threadIdx.x;
    const int lane = t & 31;
    const int w    = t >> 5;       // wave id 0..31
    const int wm   = w >> 2;       // 0..7  (32-row wave tiles)
    const int wn   = w & 3;        // 0..3  (64-col wave tiles)
    const int m0   = blockIdx.x * MB;
    const int fblk = blockIdx.y;
    const int f0   = fblk * FT;

    // init per-row top-64 lists
    for (int idx = t; idx < MB * TOPK; idx += 1024) { lv[idx] = NEG_INF; li[idx] = 0; }
    if (t < MB) { lmin[t] = NEG_INF; lpos[t] = 0; }
    __syncthreads();

    const int ln15 = lane & 15;
    const int kbA  = (lane < 16) ? 0 : 8;    // A fragment K base (16-bit A layout)
    const int kbB  = (lane < 16) ? 0 : 8;    // B fragment K base (same layout as A)
    const int rhi  = (lane >> 4) << 3;       // C layout: lanes 16..31 hold rows +8

    for (int sub = 0; sub < FT / NSUB; ++sub) {
        const int f0s = f0 + sub * NSUB;

        v8f acc[2][4];
        v8f vz = {};
        #pragma unroll
        for (int ms = 0; ms < 2; ++ms)
            #pragma unroll
            for (int ns = 0; ns < 4; ++ns) acc[ms][ns] = vz;

        // K loop, double-buffered LDS panels
        load_xpanel(xbf, xs, m0, 0, t);
        load_wpanel(encw, wsp, f0s, 0, t);
        for (int kt = 0; kt < En / KT; ++kt) {
            __syncthreads();
            const int cur = kt & 1;
            if (kt + 1 < En / KT) {
                load_xpanel(xbf, xs + ((cur ^ 1) * XS_ELEMS), m0, (kt + 1) * KT, t);
                load_wpanel(encw, wsp + ((cur ^ 1) * WS_ELEMS), f0s, (kt + 1) * KT, t);
            }
            const __bf16* xsb = xs + cur * XS_ELEMS;
            const __bf16* wsb = wsp + cur * WS_ELEMS;

            v16bf afr[2];
            #pragma unroll
            for (int ms = 0; ms < 2; ++ms) {
                const __bf16* xr = xsb + (wm * 32 + ms * 16 + ln15) * XSTR;
                v8bf lo = *(const v8bf*)(xr + kbA);
                v8bf hi = *(const v8bf*)(xr + kbA + 16);
                afr[ms] = __builtin_shufflevector(lo, hi, 0, 1, 2, 3, 4, 5, 6, 7,
                                                  8, 9, 10, 11, 12, 13, 14, 15);
            }
            #pragma unroll
            for (int ns = 0; ns < 4; ++ns) {
                const __bf16* wr = wsb + (wn * 64 + ns * 16 + ln15) * XSTR;
                v8bf lo = *(const v8bf*)(wr + kbB);
                v8bf hi = *(const v8bf*)(wr + kbB + 16);
                v16bf bfr = __builtin_shufflevector(lo, hi, 0, 1, 2, 3, 4, 5, 6, 7,
                                                    8, 9, 10, 11, 12, 13, 14, 15);
                #pragma unroll
                for (int ms = 0; ms < 2; ++ms)
                    acc[ms][ns] = __builtin_amdgcn_wmma_f32_16x16x32_bf16(
                        false, afr[ms], false, bfr, (short)0, acc[ms][ns], false, false);
            }
        }
        __syncthreads();

        // 4 phases: stage one 16-col slice per wave, then threshold-insert
        for (int p = 0; p < 4; ++p) {
            #pragma unroll
            for (int ms = 0; ms < 2; ++ms)
                #pragma unroll
                for (int r = 0; r < 8; ++r) {
                    int row = wm * 32 + ms * 16 + r + rhi;
                    stage[row * 64 + wn * 16 + ln15] = acc[ms][p][r];
                }
            __syncthreads();

            const int fbase = f0s + p * 16;
            for (int i = 0; i < 8; ++i) {
                const int r = w * 8 + i;            // each wave owns 8 rows
                float curmin = lmin[r];
                int   curpos = lpos[r];
                #pragma unroll
                for (int g = 0; g < 2; ++g) {
                    int c = g * 32 + lane;
                    float v  = stage[r * 64 + c];
                    int fidx = fbase + ((c >> 4) << 6) + (c & 15);
                    unsigned long long mk = __ballot(v > curmin);
                    while (mk) {
                        int srcl = __ffsll(mk) - 1;
                        mk &= mk - 1;
                        float vv = __shfl(v, srcl);
                        int   ff = __shfl(fidx, srcl);
                        if (vv > curmin) {
                            if (lane == 0) { lv[r * TOPK + curpos] = vv; li[r * TOPK + curpos] = ff; }
                            __builtin_amdgcn_wave_barrier();
                            float m1 = lv[r * TOPK + lane];      int p1 = lane;
                            float m2 = lv[r * TOPK + 32 + lane];
                            if (m2 < m1) { m1 = m2; p1 = lane + 32; }
                            #pragma unroll
                            for (int off = 16; off > 0; off >>= 1) {
                                float om = __shfl_xor(m1, off);
                                int   op = __shfl_xor(p1, off);
                                if (om < m1) { m1 = om; p1 = op; }
                            }
                            curmin = m1; curpos = p1;
                        }
                    }
                }
                if (lane == 0) { lmin[r] = curmin; lpos[r] = curpos; }
                __builtin_amdgcn_wave_barrier();
            }
            __syncthreads();
        }
    }

    // write partial top-64 lists: pv/pi[(row * NCHUNK + fblk) * 64 + k]
    for (int idx = t; idx < MB * TOPK; idx += 1024) {
        int r = idx >> 6, k = idx & 63;
        size_t o = ((size_t)(m0 + r) * NCHUNK + fblk) * TOPK + k;
        pv[o] = lv[idx];
        pi[o] = li[idx];
    }
}

// ======================================================================
// Kernel 2a: zero the bincount region of d_out
// ======================================================================
__global__ __launch_bounds__(256) void zero_total_kernel(int* __restrict__ total) {
    total[blockIdx.x * 256 + threadIdx.x] = 0;
}

// ======================================================================
// Kernel 2b: merge 32 x 64 partial candidates per row -> final top-64,
//            plus scatter-add bincount. One wave per row.
// ======================================================================
__global__ __launch_bounds__(256) void merge_topk_kernel(const float* __restrict__ pv,
                                                         const int* __restrict__ pi,
                                                         float* __restrict__ wsel,
                                                         int* __restrict__ fsel,
                                                         int* __restrict__ total) {
    extern __shared__ char sm[];
    const int w = threadIdx.x >> 5, lane = threadIdx.x & 31;
    float* mv = (float*)sm + w * CAND;
    int*   mi = (int*)(sm + 8 * CAND * 4) + w * CAND;
    const int row = blockIdx.x * 8 + w;

    for (int j = 0; j < CAND / 32; ++j) {
        int s = j * 32 + lane;
        mv[s] = pv[(size_t)row * CAND + s];
        mi[s] = pi[(size_t)row * CAND + s];
    }
    __builtin_amdgcn_wave_barrier();

    for (int it = 0; it < TOPK; ++it) {
        float bm = NEG_INF; int bp = lane;
        for (int j = 0; j < CAND / 32; ++j) {
            int s = j * 32 + lane;
            float vv = mv[s];
            if (vv > bm) { bm = vv; bp = s; }
        }
        #pragma unroll
        for (int off = 16; off > 0; off >>= 1) {
            float ov = __shfl_xor(bm, off);
            int   op = __shfl_xor(bp, off);
            if (ov > bm) { bm = ov; bp = op; }
        }
        int sel = __shfl(bp, 0);          // lane0's argmax: globally consistent
        if (lane == (sel & 31)) mv[sel] = NEG_INF;
        if (lane == 0) {
            wsel[row * TOPK + it] = bm;   // order irrelevant: decoder sums
            int f = mi[sel];
            fsel[row * TOPK + it] = f;
            atomicAdd(total + f, 1);
        }
        __builtin_amdgcn_wave_barrier();
    }
}

// ======================================================================
// Kernel 3: decoder gather + weighted combine + bias. One block per row.
// ======================================================================
__global__ __launch_bounds__(256) void decode_kernel(const float* __restrict__ wsel,
                                                     const int* __restrict__ fsel,
                                                     const float* __restrict__ dec,
                                                     const float* __restrict__ bias,
                                                     float* __restrict__ out) {
    __shared__ float wk[TOPK];
    __shared__ int   fk[TOPK];
    const int b = blockIdx.x, t = threadIdx.x;
    if (t < TOPK) { wk[t] = wsel[b * TOPK + t]; fk[t] = fsel[b * TOPK + t]; }
    __syncthreads();

    const int e0 = t * 8;
    float a0 = 0.f, a1 = 0.f, a2 = 0.f, a3 = 0.f, a4 = 0.f, a5 = 0.f, a6 = 0.f, a7 = 0.f;
    for (int k = 0; k < TOPK; ++k) {
        float wv = wk[k];
        const float4* dr = (const float4*)(dec + (size_t)fk[k] * En + e0);
        float4 d0 = dr[0], d1 = dr[1];
        a0 += wv * d0.x; a1 += wv * d0.y; a2 += wv * d0.z; a3 += wv * d0.w;
        a4 += wv * d1.x; a5 += wv * d1.y; a6 += wv * d1.z; a7 += wv * d1.w;
    }
    const float4* b4 = (const float4*)(bias + e0);
    float4 bb0 = b4[0], bb1 = b4[1];
    float4 o0 = {a0 + bb0.x, a1 + bb0.y, a2 + bb0.z, a3 + bb0.w};
    float4 o1 = {a4 + bb1.x, a5 + bb1.y, a6 + bb1.z, a7 + bb1.w};
    float4* ob = (float4*)(out + (size_t)b * En + e0);
    ob[0] = o0; ob[1] = o1;
}

// ======================================================================
// launch
// ======================================================================
extern "C" void kernel_launch(void* const* d_in, const int* in_sizes, int n_in,
                              void* d_out, int out_size, void* d_ws, size_t ws_size,
                              hipStream_t stream) {
    const float* embed = (const float*)d_in[0];   // [B, E]
    const float* bias  = (const float*)d_in[1];   // [E]
    const float* encw  = (const float*)d_in[2];   // [F, E]
    const float* dec   = (const float*)d_in[3];   // [F, E]

    float* out   = (float*)d_out;                 // embed1: B*E floats
    int*   total = (int*)(out + (size_t)Bn * En); // bincount: F int32

    char* ws = (char*)d_ws;
    __bf16* xbf = (__bf16*)(ws);                                  // 16 MB
    float*  pv  = (float*)(ws + (size_t)Bn * En * 2);             // 32 MB
    int*    pi  = (int*)((char*)pv + (size_t)Bn * CAND * 4);      // 32 MB
    float*  wsel = (float*)((char*)pi + (size_t)Bn * CAND * 4);   // 1 MB
    int*    fsel = (int*)((char*)wsel + (size_t)Bn * TOPK * 4);   // 1 MB

    (void)in_sizes; (void)n_in; (void)out_size; (void)ws_size;

    hipFuncSetAttribute(reinterpret_cast<const void*>(gemm_topk_kernel),
                        hipFuncAttributeMaxDynamicSharedMemorySize, SMEM_GEMM);
    hipFuncSetAttribute(reinterpret_cast<const void*>(merge_topk_kernel),
                        hipFuncAttributeMaxDynamicSharedMemorySize, SMEM_MERGE);

    // 0: bias-subtract + bf16 convert of activations
    prep_x_kernel<<<(Bn * En) / (256 * 8), 256, 0, stream>>>(embed, bias, xbf);

    // 1: fused WMMA GEMM + per-chunk top-64 (f-chunk-major grid => enc_w slice L2-resident)
    gemm_topk_kernel<<<dim3(Bn / MB, NCHUNK), 1024, SMEM_GEMM, stream>>>(xbf, encw, pv, pi);

    // 2: bincount zero + candidate merge (+ scatter-add bincount)
    zero_total_kernel<<<Fn / 256, 256, 0, stream>>>(total);
    merge_topk_kernel<<<Bn / 8, 256, SMEM_MERGE, stream>>>(pv, pi, wsel, fsel, total);

    // 3: decoder gather (HBM-bound, ~2 GB coalesced)
    decode_kernel<<<Bn, 256, 0, stream>>>(wsel, fsel, dec, bias, out);
}
